// Conv2d_53901839564892
// MI455X (gfx1250) — matrix operand order repaired and checked by reading, not compiled, not run on hardware
//
#include <hip/hip_runtime.h>

// CDNA5 / gfx1250 types for WMMA
typedef __attribute__((ext_vector_type(16))) __bf16 v16bf;
typedef __attribute__((ext_vector_type(8)))  float  v8f;

#define CIN   32
#define HIMG  128
#define WIMG  128
#define OOUT  64
#define CP    40            // LDS channel stride in ushorts (80 B rows: 16B-aligned)
#define PLANE (20 * CP)     // ushorts per (slot, hi/lo) plane = 800 (1600 B)
#define SLOT  (2 * PLANE)   // ushorts per row slot (hi plane + lo plane) = 1600

union AFrag { v16bf v; uint4 q[2]; };
union BFrag { v16bf v; unsigned u[8]; };

// pack two fp32 into (bf16(f0) | bf16(f1)<<16), truncating
static __device__ __forceinline__ unsigned pack_hi(unsigned u0, unsigned u1) {
    return (u0 >> 16) | (u1 & 0xFFFF0000u);
}

// Load one input row (gh) for this tile into LDS slot `slotbase`:
// 288 channel-pairs, split fp32 -> bf16 hi/lo once per element.
static __device__ __forceinline__ void fill_row(const float* __restrict__ x,
                                                unsigned short* __restrict__ xs,
                                                int n, int w0, int gh,
                                                int slotbase, int tid)
{
    const bool hok = (unsigned)gh < (unsigned)HIMG;
    for (int p = tid; p < (CIN / 2) * 18; p += 128) {
        int cp = p / 18;
        int j  = p - cp * 18;
        int c  = cp * 2;
        int gw = w0 - 1 + j;
        float f0 = 0.0f, f1 = 0.0f;
        if (hok && (unsigned)gw < (unsigned)WIMG) {
            const float* px = x + (((size_t)n * CIN + c) * HIMG + gh) * WIMG + gw;
            f0 = px[0];
            f1 = px[(size_t)HIMG * WIMG];
        }
        unsigned u0 = __float_as_uint(f0), u1 = __float_as_uint(f1);
        float l0 = f0 - __uint_as_float(u0 & 0xFFFF0000u);
        float l1 = f1 - __uint_as_float(u1 & 0xFFFF0000u);
        unsigned short* d = xs + slotbase + j * CP + c;
        *(unsigned*)(d)         = pack_hi(u0, u1);
        *(unsigned*)(d + PLANE) = pack_hi(__float_as_uint(l0), __float_as_uint(l1));
    }
}

__global__ __launch_bounds__(128)
void conv3x3_wmma_bf16x3(const float* __restrict__ x,
                         const float* __restrict__ wgt,
                         float* __restrict__ out)
{
    const int tid  = threadIdx.x;
    const int lane = tid & 31;
    const int wave = tid >> 5;

    const int n  = blockIdx.x;          // batch          0..31
    const int w0 = blockIdx.y * 16;     // w tile origin  0..112
    const int h0 = blockIdx.z * 32;     // h chunk origin 0..96

    const int obase = wave * 16;        // each wave owns 16 output channels
    const int nn    = lane & 15;
    const int o     = obase + nn;

    // Ring buffer: 3 row slots x (hi,lo) planes x 20 cols x CP channels (bf16)
    __shared__ __align__(16) unsigned short xs[3 * SLOT];

    // ---------------- B fragments (weights), built once, live in VGPRs ----
    // B 32x16 bf16, contiguous k order used here: lanes 0-15 hold K=0..15 (N=lane), lanes 16-31 K=16..31.
    // Effective weight (reference reinterpretation): wgt[(c*9 + tap)*64 + o].
    const int cB = (lane < 16) ? 0 : 16;
    v16bf bhi[9], blo[9];
#pragma unroll
    for (int tap = 0; tap < 9; ++tap) {
        BFrag th, tl;
#pragma unroll
        for (int i = 0; i < 8; ++i) {          // packed channel pairs
            const int c = cB + 2 * i;
            float f0 = wgt[(c * 9 + tap) * OOUT + o];
            float f1 = wgt[((c + 1) * 9 + tap) * OOUT + o];
            unsigned u0 = __float_as_uint(f0), u1 = __float_as_uint(f1);
            float l0 = f0 - __uint_as_float(u0 & 0xFFFF0000u);
            float l1 = f1 - __uint_as_float(u1 & 0xFFFF0000u);
            th.u[i] = pack_hi(u0, u1);
            tl.u[i] = pack_hi(__float_as_uint(l0), __float_as_uint(l1));
        }
        bhi[tap] = th.v;
        blo[tap] = tl.v;
    }

    // A fragment K bases per lane group (contiguous k order used here, same as B), ushort units:
// lanes 0-15: elems 0..15 -> K=0..15
    // lanes16-31: elems 0..15 -> K=16..31
    const int cA0  = (lane < 16) ? 0 : 16;
    const int colM = lane & 15;                      // M = w offset in tile
    const unsigned short* pbase = xs + colM * CP + cA0;  // 16B-aligned per lane

    // Output: lane holds channel o; D vgpr v -> w = w0 + woff + v (8 contiguous).
    const int woff = (lane < 16) ? 0 : 8;
    float* outlane = out + (((size_t)n * OOUT + o) * HIMG) * WIMG + w0 + woff;

    // ---- prologue: prime ring buffer with rows h0-1 (slot0) and h0 (slot1) ----
    int sb0 = 0, sb1 = SLOT, sb2 = 2 * SLOT;         // slots for rows h-1, h, h+1
    fill_row(x, xs, n, w0, h0 - 1, sb0, tid);
    fill_row(x, xs, n, w0, h0,     sb1, tid);

    for (int h = h0; h < h0 + 32; ++h) {
        __syncthreads();   // prior iteration's readers done before slot reuse
        fill_row(x, xs, n, w0, h + 1, sb2, tid);     // slide window: one new row
        if (h + 2 < HIMG)  // warm next iteration's fill
            __builtin_prefetch(x + (((size_t)n * CIN + lane) * HIMG + (h + 2)) * WIMG + w0, 0, 1);
        __syncthreads();

        // per-row fragment base addresses (3 v_adds; everything else immediate)
        const unsigned short* prs[3] = { pbase + sb0, pbase + sb1, pbase + sb2 };

        // ---- software-pipelined tap loop: double-buffered A fragments ----
        AFrag hbuf[2], lbuf[2];
        {   // preload tap 0 (dy=0, dx=0)
            const unsigned short* pr = prs[0];
            hbuf[0].q[0] = *(const uint4*)(pr);
            hbuf[0].q[1] = *(const uint4*)(pr + 8);
            lbuf[0].q[0] = *(const uint4*)(pr + PLANE);
            lbuf[0].q[1] = *(const uint4*)(pr + PLANE + 8);
        }

        v8f acc = {};
#pragma unroll
        for (int tap = 0; tap < 9; ++tap) {
            const int cur = tap & 1, nxt = cur ^ 1;
            if (tap < 8) {                           // issue next tap's loads first
                const int dy1 = (tap + 1) / 3, dx1 = (tap + 1) - dy1 * 3;
                const unsigned short* pr = prs[dy1];
                hbuf[nxt].q[0] = *(const uint4*)(pr + dx1 * CP);
                hbuf[nxt].q[1] = *(const uint4*)(pr + dx1 * CP + 8);
                lbuf[nxt].q[0] = *(const uint4*)(pr + dx1 * CP + PLANE);
                lbuf[nxt].q[1] = *(const uint4*)(pr + dx1 * CP + PLANE + 8);
            }
            // fp32-via-bf16x3: hi*hi + hi*lo + lo*hi (lo*lo dropped, ~2^-16 rel)
            acc = __builtin_amdgcn_wmma_f32_16x16x32_bf16(false, hbuf[cur].v, false, bhi[tap],
                                                          (short)0, acc, false, false);
            acc = __builtin_amdgcn_wmma_f32_16x16x32_bf16(false, hbuf[cur].v, false, blo[tap],
                                                          (short)0, acc, false, false);
            acc = __builtin_amdgcn_wmma_f32_16x16x32_bf16(false, lbuf[cur].v, false, bhi[tap],
                                                          (short)0, acc, false, false);
        }

        // ---- store: 8 contiguous fp32 per lane (two b128 stores) ----
        float* po = outlane + (size_t)h * WIMG;
        *(float4*)(po)     = make_float4(acc[0], acc[1], acc[2], acc[3]);
        *(float4*)(po + 4) = make_float4(acc[4], acc[5], acc[6], acc[7]);

        // rotate ring: (h-1, h, h+1) slots advance by one row
        int t = sb0; sb0 = sb1; sb1 = sb2; sb2 = t;
    }
}

extern "C" void kernel_launch(void* const* d_in, const int* in_sizes, int n_in,
                              void* d_out, int out_size, void* d_ws, size_t ws_size,
                              hipStream_t stream) {
    const float* x   = (const float*)d_in[0];   // (32, 32, 128, 128) fp32 NCHW
    const float* wgt = (const float*)d_in[1];   // raw (3,3,32,64) buffer, read as (C,3,3,O)
    float* out       = (float*)d_out;           // (32, 64, 128, 128) fp32

    dim3 grid(32, WIMG / 16, HIMG / 32);        // (n, w-tile, h-chunk)
    dim3 block(128);                            // 4 waves: one 16-O slice each
    hipLaunchKernelGGL(conv3x3_wmma_bf16x3, grid, block, 0, stream, x, wgt, out);
}
